// crosstask_performer_652835029387
// MI455X (gfx1250) — matrix operand order repaired and checked by reading, not compiled, not run on hardware
//
#include <hip/hip_runtime.h>
#include <hip/hip_bf16.h>

typedef __attribute__((ext_vector_type(16))) _Float16 v16h;
typedef __attribute__((ext_vector_type(8)))  _Float16 v8h;
typedef __attribute__((ext_vector_type(8)))  float    v8f;
typedef int v4i_gcc __attribute__((vector_size(16)));   // matches async-LDS builtin param

#define GEMM_LDSS 72   // 144B row stride: b128 frag reads conflict-free (36m mod 64 distinct)
#define KPTV_LDSS 40   // 80B row stride (BK=32 path)

#define GLOBAL_AS __attribute__((address_space(1)))
#define SHARED_AS __attribute__((address_space(3)))

#if defined(__gfx1250__) && __has_builtin(__builtin_amdgcn_global_load_async_to_lds_b128)
#define HAVE_ASYNC_LDS 1
#else
#define HAVE_ASYNC_LDS 0
#endif

#if defined(__gfx1250__) && __has_builtin(__builtin_amdgcn_s_wait_asynccnt)
#define ASYNC_WAIT(n) __builtin_amdgcn_s_wait_asynccnt(n)
#else
#define ASYNC_WAIT(n)
#endif

// ---------------------------------------------------------------------------
// Fragment loaders from LDS (tile stored as [row][k], row stride in halfs)
// A layout (16x32 f16): lanes0-15 row m, V0-3 = k0..7, V4-7 = k16..23 (k0=8*(lane>>4))
// B layout (32x16 f16): lane -> col n=lane&15, V0-3 = k0..7, V4-7 = k16..23 (k0=8*(lane>>4))
// ---------------------------------------------------------------------------
__device__ __forceinline__ v16h load_frag_A(const _Float16* s, int stride, int rowBase, int lane) {
    const int m  = rowBase + (lane & 15);
    const int kh = lane >> 4;
    union { v16h v; v8h h[2]; } u;
    u.h[0] = *(const v8h*)(s + m * stride + 8 * kh);
    u.h[1] = *(const v8h*)(s + m * stride + 16 + 8 * kh);
    return u.v;
}
__device__ __forceinline__ v16h load_frag_B(const _Float16* s, int stride, int colBase, int lane) {
    const int n  = colBase + (lane & 15);
    const int k0 = (lane >> 4) * 8;
    union { v16h v; v8h h[2]; } u;
    u.h[0] = *(const v8h*)(s + n * stride + k0);
    u.h[1] = *(const v8h*)(s + n * stride + k0 + 16);
    return u.v;
}

// ---------------------------------------------------------------------------
// Generic WMMA GEMM: out[r,n] = epilogue( sum_k A[r,k]*W[n,k] )
// A: [nbatch*rows, K] f16,  W: [N, K] f16 (+ per-batch stride).
// 128x128 block tile, BK=64 (2 WMMA k-steps per stage), 8 waves (2M x 4N),
// each wave 64x32 via 4x2 v_wmma_f32_16x16x32_f16.
// Double-buffered LDS; stages copied with GLOBAL_LOAD_ASYNC_TO_LDS_B128
// (ASYNCcnt pipeline: 8 async ops/wave/stage, s_wait_asynccnt 8) when the
// toolchain declares the builtin, else synchronous v8h copies.
// EPI: 0 = +bias -> f16                      (QKV)
//      1 = exp(acc-aux[row])*scale -> f16    (Performer feature map)
//      2 = acc/(aux[row]+1e-8) -> f16        (y numerator)
//      3 = acc+bias+resid[row,col] -> f32    (proj / mlp2, residual add)
//      4 = gelu(acc+bias) -> f16             (mlp1)
// ---------------------------------------------------------------------------
template <int EPI>
__global__ void __launch_bounds__(256)
gemm_wmma_kernel(const _Float16* __restrict__ A, const _Float16* __restrict__ W,
                 void* __restrict__ outv, const float* __restrict__ bias,
                 const float* __restrict__ aux, const float* __restrict__ resid,
                 int rows, int N, int K, long wBatchStride, float scale)
{
    const int z = blockIdx.z;
    A += (size_t)z * rows * K;
    W += (size_t)z * wBatchStride;
    const size_t obase = (size_t)z * rows * N;
    if (aux)   aux   += (size_t)z * rows;
    if (resid) resid += (size_t)z * rows * (size_t)N;
    _Float16* outH = (_Float16*)outv + obase;
    float*    outF = (float*)outv + obase;

    __shared__ _Float16 As[2][128 * GEMM_LDSS];
    __shared__ _Float16 Bs[2][128 * GEMM_LDSS];

    const int tid  = threadIdx.x;
    const int lane = tid & 31;
    const int wave = tid >> 5;
    const int wm   = wave & 1;   // 2 waves along M
    const int wn   = wave >> 1;  // 4 waves along N
    const int blockM = blockIdx.y * 128;
    const int blockN = blockIdx.x * 128;

    v8f acc[4][2];
    const v8f z8 = {0.f, 0.f, 0.f, 0.f, 0.f, 0.f, 0.f, 0.f};
#pragma unroll
    for (int i = 0; i < 4; ++i)
#pragma unroll
        for (int j = 0; j < 2; ++j) acc[i][j] = z8;

    const int lrow = tid >> 1;         // 0..127  (tile row)
    const int lseg = (tid & 1) * 32;   // 0 or 32 (k offset, halfs)

    auto issue_stage = [&](int buf, int kt) {
        const _Float16* ga = A + (size_t)(blockM + lrow) * K + kt + lseg;
        const _Float16* gb = W + (size_t)(blockN + lrow) * K + kt + lseg;
        _Float16* la = &As[buf][lrow * GEMM_LDSS + lseg];
        _Float16* lb = &Bs[buf][lrow * GEMM_LDSS + lseg];
#if HAVE_ASYNC_LDS
#pragma unroll
        for (int c = 0; c < 32; c += 8) {
            __builtin_amdgcn_global_load_async_to_lds_b128(
                (GLOBAL_AS v4i_gcc*)(ga + c), (SHARED_AS v4i_gcc*)(la + c), 0, 0);
            __builtin_amdgcn_global_load_async_to_lds_b128(
                (GLOBAL_AS v4i_gcc*)(gb + c), (SHARED_AS v4i_gcc*)(lb + c), 0, 0);
        }
#else
#pragma unroll
        for (int c = 0; c < 32; c += 8) {
            *(v8h*)(la + c) = *(const v8h*)(ga + c);
            *(v8h*)(lb + c) = *(const v8h*)(gb + c);
        }
#endif
    };

    const int nk = K >> 6;   // BK = 64
    issue_stage(0, 0);
    for (int i = 0; i < nk; ++i) {
        const int cur = i & 1;
        if (i + 1 < nk) issue_stage(cur ^ 1, (i + 1) << 6);
#if HAVE_ASYNC_LDS
        if (i + 1 < nk) { ASYNC_WAIT(8); } else { ASYNC_WAIT(0); }
#endif
        __syncthreads();
#pragma unroll
        for (int kk = 0; kk < 2; ++kk) {
            const _Float16* as = &As[cur][kk * 32];
            const _Float16* bs = &Bs[cur][kk * 32];
            v16h af[4], bf[2];
#pragma unroll
            for (int im = 0; im < 4; ++im)
                af[im] = load_frag_A(as, GEMM_LDSS, wm * 64 + im * 16, lane);
#pragma unroll
            for (int in_ = 0; in_ < 2; ++in_)
                bf[in_] = load_frag_B(bs, GEMM_LDSS, wn * 32 + in_ * 16, lane);
#pragma unroll
            for (int im = 0; im < 4; ++im)
#pragma unroll
                for (int in_ = 0; in_ < 2; ++in_)
                    acc[im][in_] = __builtin_amdgcn_wmma_f32_16x16x32_f16(
                        false, af[im], false, bf[in_], (short)0, acc[im][in_], false, false);
        }
        __syncthreads();
    }

    const int lhalf = lane >> 4;
    const int lcol  = lane & 15;
#pragma unroll
    for (int in_ = 0; in_ < 2; ++in_) {
        const int col = blockN + wn * 32 + in_ * 16 + lcol;
        float bv = 0.f;
        if constexpr (EPI == 0 || EPI == 3 || EPI == 4) bv = bias[col];
#pragma unroll
        for (int im = 0; im < 4; ++im) {
            const int rbase = blockM + wm * 64 + im * 16 + 8 * lhalf;
#pragma unroll
            for (int v = 0; v < 8; ++v) {
                const int row = rbase + v;
                const float x = acc[im][in_][v];
                if constexpr (EPI == 0) {
                    outH[(size_t)row * N + col] = (_Float16)(x + bv);
                } else if constexpr (EPI == 1) {
                    outH[(size_t)row * N + col] = (_Float16)(expf(x - aux[row]) * scale);
                } else if constexpr (EPI == 2) {
                    outH[(size_t)row * N + col] = (_Float16)(x / (aux[row] + 1e-8f));
                } else if constexpr (EPI == 3) {
                    outF[(size_t)row * N + col] = x + bv + resid[(size_t)row * N + col];
                } else {
                    const float t = x + bv;
                    outH[(size_t)row * N + col] =
                        (_Float16)(0.5f * t * (1.0f + erff(t * 0.70710678f)));
                }
            }
        }
    }
    (void)scale;
}

// ---------------------------------------------------------------------------
// kptv[b,n,m] = sum_t V[b,t,n] * KP[b,t,m]   (K = T; both operands K-major in
// global, loader writes transposed into LDS, then same WMMA core, BK=32)
// ---------------------------------------------------------------------------
__global__ void __launch_bounds__(256)
kptv_wmma_kernel(const _Float16* __restrict__ V, const _Float16* __restrict__ KP,
                 _Float16* __restrict__ out, int T, int E, int M)
{
    const int z = blockIdx.z;
    V   += (size_t)z * T * E;
    KP  += (size_t)z * T * M;
    out += (size_t)z * E * M;

    __shared__ _Float16 As[128 * KPTV_LDSS];  // [n_local][t_local]
    __shared__ _Float16 Bs[128 * KPTV_LDSS];  // [m_local][t_local]

    const int tid  = threadIdx.x;
    const int lane = tid & 31;
    const int wave = tid >> 5;
    const int wm   = wave & 1;
    const int wn   = wave >> 1;
    const int nBase = blockIdx.y * 128;
    const int mBase = blockIdx.x * 128;

    v8f acc[4][2];
    const v8f z8 = {0.f, 0.f, 0.f, 0.f, 0.f, 0.f, 0.f, 0.f};
#pragma unroll
    for (int i = 0; i < 4; ++i)
#pragma unroll
        for (int j = 0; j < 2; ++j) acc[i][j] = z8;

    const int tl0 = tid >> 4;        // 0..15
    const int c8  = (tid & 15) * 8;  // 0..120

    for (int kt = 0; kt < T; kt += 32) {
#pragma unroll
        for (int rep = 0; rep < 2; ++rep) {
            const int tl = tl0 + rep * 16;
            const v8h vv = *(const v8h*)(V  + (size_t)(kt + tl) * E + nBase + c8);
            const v8h kk = *(const v8h*)(KP + (size_t)(kt + tl) * M + mBase + c8);
#pragma unroll
            for (int j = 0; j < 8; ++j) {
                As[(c8 + j) * KPTV_LDSS + tl] = vv[j];
                Bs[(c8 + j) * KPTV_LDSS + tl] = kk[j];
            }
        }
        __syncthreads();

        v16h af[4], bf[2];
#pragma unroll
        for (int im = 0; im < 4; ++im)
            af[im] = load_frag_A(As, KPTV_LDSS, wm * 64 + im * 16, lane);
#pragma unroll
        for (int in_ = 0; in_ < 2; ++in_)
            bf[in_] = load_frag_B(Bs, KPTV_LDSS, wn * 32 + in_ * 16, lane);
#pragma unroll
        for (int im = 0; im < 4; ++im)
#pragma unroll
            for (int in_ = 0; in_ < 2; ++in_)
                acc[im][in_] = __builtin_amdgcn_wmma_f32_16x16x32_f16(
                    false, af[im], false, bf[in_], (short)0, acc[im][in_], false, false);
        __syncthreads();
    }

    const int lhalf = lane >> 4;
    const int lcol  = lane & 15;
#pragma unroll
    for (int in_ = 0; in_ < 2; ++in_) {
        const int col = mBase + wn * 32 + in_ * 16 + lcol;
#pragma unroll
        for (int im = 0; im < 4; ++im) {
            const int rbase = nBase + wm * 64 + im * 16 + 8 * lhalf;
#pragma unroll
            for (int v = 0; v < 8; ++v)
                out[(size_t)(rbase + v) * M + col] = (_Float16)acc[im][in_][v];
        }
    }
}

// --------------------------- small helper kernels --------------------------
__global__ void __launch_bounds__(256)
layernorm_f16_kernel(const float* __restrict__ x, const float* __restrict__ g,
                     const float* __restrict__ b, _Float16* __restrict__ y, int D)
{
    const size_t row = blockIdx.x;
    x += row * D; y += row * D;
    __shared__ float r1[256], r2[256];
    float s = 0.f, s2 = 0.f;
    for (int i = threadIdx.x; i < D; i += 256) { const float v = x[i]; s += v; s2 += v * v; }
    r1[threadIdx.x] = s; r2[threadIdx.x] = s2;
    __syncthreads();
    for (int o = 128; o > 0; o >>= 1) {
        if (threadIdx.x < o) { r1[threadIdx.x] += r1[threadIdx.x + o];
                               r2[threadIdx.x] += r2[threadIdx.x + o]; }
        __syncthreads();
    }
    const float mu   = r1[0] / D;
    const float var  = r2[0] / D - mu * mu;
    const float rstd = rsqrtf(var + 1e-5f);
    for (int i = threadIdx.x; i < D; i += 256)
        y[i] = (_Float16)((x[i] - mu) * rstd * g[i] + b[i]);
}

__global__ void __launch_bounds__(256)
rowsumsq_kernel(const _Float16* __restrict__ x, float* __restrict__ out, int K)
{
    const int row  = blockIdx.x * 8 + (threadIdx.x >> 5);
    const int lane = threadIdx.x & 31;
    const _Float16* p = x + (size_t)row * K;
    float s = 0.f;
    for (int i = lane; i < K; i += 32) { const float v = (float)p[i]; s += v * v; }
    for (int o = 16; o > 0; o >>= 1) s += __shfl_xor(s, o, 32);
    if (lane == 0) out[row] = 0.5f * s;
}

__global__ void __launch_bounds__(256)
rowdot_kernel(const _Float16* __restrict__ qp, const float* __restrict__ ksum,
              float* __restrict__ out, int M, int T)
{
    const int row  = blockIdx.x * 8 + (threadIdx.x >> 5);
    const int lane = threadIdx.x & 31;
    const int bb   = row / T;
    const _Float16* p = qp + (size_t)row * M;
    const float* ks = ksum + (size_t)bb * M;
    float s = 0.f;
    for (int i = lane; i < M; i += 32) s += (float)p[i] * ks[i];
    for (int o = 16; o > 0; o >>= 1) s += __shfl_xor(s, o, 32);
    if (lane == 0) out[row] = s;
}

__global__ void __launch_bounds__(256)
colsum_kernel(const _Float16* __restrict__ kp, float* __restrict__ out, int T, int M)
{
    const int bb = blockIdx.y;
    const int m  = blockIdx.x * blockDim.x + threadIdx.x;
    if (m >= M) return;
    const _Float16* p = kp + (size_t)bb * T * M + m;
    float s = 0.f;
    for (int t = 0; t < T; ++t) s += (float)p[(size_t)t * M];
    out[(size_t)bb * M + m] = s;
}

__global__ void __launch_bounds__(256)
f32_to_f16_kernel(const float* __restrict__ in, _Float16* __restrict__ out, int n)
{
    const int i = blockIdx.x * blockDim.x + threadIdx.x;
    if (i < n) out[i] = (_Float16)in[i];
}

// ---------------------------------------------------------------------------
extern "C" void kernel_launch(void* const* d_in, const int* in_sizes, int n_in,
                              void* d_out, int out_size, void* d_ws, size_t ws_size,
                              hipStream_t stream)
{
    (void)in_sizes; (void)n_in; (void)out_size; (void)ws_size;
    constexpr int Bb = 8, Tt = 4096, Dd = 512, Ee = 512, Mm = 256;
    constexpr size_t Rr = (size_t)Bb * Tt;   // 32768 rows

    const float* saliency = (const float*)d_in[0];
    const float* contour  = (const float*)d_in[1];

    char* wsb = (char*)d_ws;
    size_t off = 0;
    auto alloc = [&](size_t bytes) -> char* {
        char* p = wsb + off; off += (bytes + 255) & ~(size_t)255; return p;
    };

    // f16 weights
    _Float16* h_qs  = (_Float16*)alloc((size_t)Ee * Dd * 2);
    _Float16* h_ks  = (_Float16*)alloc((size_t)Ee * Dd * 2);
    _Float16* h_vs  = (_Float16*)alloc((size_t)Ee * Dd * 2);
    _Float16* h_qc  = (_Float16*)alloc((size_t)Ee * Dd * 2);
    _Float16* h_kc  = (_Float16*)alloc((size_t)Ee * Dd * 2);
    _Float16* h_vc  = (_Float16*)alloc((size_t)Ee * Dd * 2);
    _Float16* h_ps  = (_Float16*)alloc((size_t)Ee * Ee * 2);
    _Float16* h_pc  = (_Float16*)alloc((size_t)Ee * Ee * 2);
    _Float16* h_m1s = (_Float16*)alloc((size_t)Ee * Ee * 2);
    _Float16* h_m2s = (_Float16*)alloc((size_t)Ee * Ee * 2);
    _Float16* h_m1c = (_Float16*)alloc((size_t)Ee * Ee * 2);
    _Float16* h_m2c = (_Float16*)alloc((size_t)Ee * Ee * 2);
    _Float16* h_wsm = (_Float16*)alloc((size_t)Mm * Ee * 2);
    _Float16* h_wcm = (_Float16*)alloc((size_t)Mm * Ee * 2);

    // activations
    _Float16* ns_h = (_Float16*)alloc(Rr * Dd * 2);   // reused as ynum_s
    _Float16* nc_h = (_Float16*)alloc(Rr * Dd * 2);   // reused as ynum_c
    _Float16* q_s  = (_Float16*)alloc(Rr * Ee * 2);   // reused as ln2s
    _Float16* k_s  = (_Float16*)alloc(Rr * Ee * 2);   // reused as ln2c
    _Float16* v_s  = (_Float16*)alloc(Rr * Ee * 2);   // reused as h_s (gelu)
    _Float16* q_c  = (_Float16*)alloc(Rr * Ee * 2);   // reused as h_c (gelu)
    _Float16* k_c  = (_Float16*)alloc(Rr * Ee * 2);
    _Float16* v_c  = (_Float16*)alloc(Rr * Ee * 2);
    _Float16* qp_s = (_Float16*)alloc(Rr * Mm * 2);
    _Float16* kp_s = (_Float16*)alloc(Rr * Mm * 2);
    _Float16* qp_c = (_Float16*)alloc(Rr * Mm * 2);
    _Float16* kp_c = (_Float16*)alloc(Rr * Mm * 2);
    float* xd_qs = (float*)alloc(Rr * 4);
    float* xd_kc = (float*)alloc(Rr * 4);
    float* xd_qc = (float*)alloc(Rr * 4);
    float* xd_ks = (float*)alloc(Rr * 4);
    float* ksum_s = (float*)alloc((size_t)Bb * Mm * 4);
    float* ksum_c = (float*)alloc((size_t)Bb * Mm * 4);
    _Float16* kptv_s = (_Float16*)alloc((size_t)Bb * Ee * Mm * 2);
    _Float16* kptv_c = (_Float16*)alloc((size_t)Bb * Ee * Mm * 2);
    float* Dsv = (float*)alloc(Rr * 4);
    float* Dcv = (float*)alloc(Rr * 4);
    float* s_pre = (float*)alloc(Rr * Ee * 4);
    float* c_pre = (float*)alloc(Rr * Ee * 4);

    _Float16* ynum_s = ns_h;  _Float16* ynum_c = nc_h;
    _Float16* ln2s = q_s;     _Float16* ln2c = k_s;
    _Float16* gel_s = v_s;    _Float16* gel_c = q_c;

    auto conv = [&](int idx, _Float16* dst, int n) {
        f32_to_f16_kernel<<<dim3((n + 255) / 256), dim3(256), 0, stream>>>(
            (const float*)d_in[idx], dst, n);
    };
    conv(2, h_qs, Ee * Dd);  conv(4, h_ks, Ee * Dd);  conv(6, h_vs, Ee * Dd);
    conv(8, h_qc, Ee * Dd);  conv(10, h_kc, Ee * Dd); conv(12, h_vc, Ee * Dd);
    conv(14, h_ps, Ee * Ee); conv(16, h_pc, Ee * Ee);
    conv(26, h_m1s, Ee * Ee); conv(28, h_m2s, Ee * Ee);
    conv(30, h_m1c, Ee * Ee); conv(32, h_m2c, Ee * Ee);
    conv(34, h_wsm, Mm * Ee); conv(35, h_wcm, Mm * Ee);

    auto gemm = [&](int epi, const _Float16* A, const _Float16* W, void* out,
                    const float* bias, const float* aux, const float* resid,
                    int rows, int N, int K, int nb, long wBatch, float scale) {
        dim3 g(N / 128, rows / 128, nb), blk(256);
        switch (epi) {
        case 0: gemm_wmma_kernel<0><<<g, blk, 0, stream>>>(A, W, out, bias, aux, resid, rows, N, K, wBatch, scale); break;
        case 1: gemm_wmma_kernel<1><<<g, blk, 0, stream>>>(A, W, out, bias, aux, resid, rows, N, K, wBatch, scale); break;
        case 2: gemm_wmma_kernel<2><<<g, blk, 0, stream>>>(A, W, out, bias, aux, resid, rows, N, K, wBatch, scale); break;
        case 3: gemm_wmma_kernel<3><<<g, blk, 0, stream>>>(A, W, out, bias, aux, resid, rows, N, K, wBatch, scale); break;
        default: gemm_wmma_kernel<4><<<g, blk, 0, stream>>>(A, W, out, bias, aux, resid, rows, N, K, wBatch, scale); break;
        }
    };

    // 1) LayerNorm 1 -> f16
    layernorm_f16_kernel<<<(unsigned)Rr, 256, 0, stream>>>(
        saliency, (const float*)d_in[18], (const float*)d_in[19], ns_h, Dd);
    layernorm_f16_kernel<<<(unsigned)Rr, 256, 0, stream>>>(
        contour, (const float*)d_in[20], (const float*)d_in[21], nc_h, Dd);

    // 2) QKV projections (+bias)
    gemm(0, ns_h, h_qs, q_s, (const float*)d_in[3], nullptr, nullptr, (int)Rr, Ee, Dd, 1, 0, 0.f);
    gemm(0, ns_h, h_ks, k_s, (const float*)d_in[5], nullptr, nullptr, (int)Rr, Ee, Dd, 1, 0, 0.f);
    gemm(0, ns_h, h_vs, v_s, (const float*)d_in[7], nullptr, nullptr, (int)Rr, Ee, Dd, 1, 0, 0.f);
    gemm(0, nc_h, h_qc, q_c, (const float*)d_in[9], nullptr, nullptr, (int)Rr, Ee, Dd, 1, 0, 0.f);
    gemm(0, nc_h, h_kc, k_c, (const float*)d_in[11], nullptr, nullptr, (int)Rr, Ee, Dd, 1, 0, 0.f);
    gemm(0, nc_h, h_vc, v_c, (const float*)d_in[13], nullptr, nullptr, (int)Rr, Ee, Dd, 1, 0, 0.f);

    // 3) xd = 0.5*|x|^2 per row
    rowsumsq_kernel<<<(unsigned)(Rr / 8), 256, 0, stream>>>(q_s, xd_qs, Ee);
    rowsumsq_kernel<<<(unsigned)(Rr / 8), 256, 0, stream>>>(k_c, xd_kc, Ee);
    rowsumsq_kernel<<<(unsigned)(Rr / 8), 256, 0, stream>>>(q_c, xd_qc, Ee);
    rowsumsq_kernel<<<(unsigned)(Rr / 8), 256, 0, stream>>>(k_s, xd_ks, Ee);

    // 4) Performer feature maps: exp(w x - xd)/sqrt(M)
    const float rsM = 0.0625f;  // 1/sqrt(256)
    gemm(1, q_s, h_wsm, qp_s, nullptr, xd_qs, nullptr, (int)Rr, Mm, Ee, 1, 0, rsM);
    gemm(1, k_c, h_wsm, kp_s, nullptr, xd_kc, nullptr, (int)Rr, Mm, Ee, 1, 0, rsM);
    gemm(1, q_c, h_wcm, qp_c, nullptr, xd_qc, nullptr, (int)Rr, Mm, Ee, 1, 0, rsM);
    gemm(1, k_s, h_wcm, kp_c, nullptr, xd_ks, nullptr, (int)Rr, Mm, Ee, 1, 0, rsM);

    // 5) ksum[b,m] = sum_t kp
    colsum_kernel<<<dim3(1, Bb), 256, 0, stream>>>(kp_s, ksum_s, Tt, Mm);
    colsum_kernel<<<dim3(1, Bb), 256, 0, stream>>>(kp_c, ksum_c, Tt, Mm);

    // 6) kptv[b,n,m] = sum_t v[b,t,n] kp[b,t,m]
    kptv_wmma_kernel<<<dim3(Mm / 128, Ee / 128, Bb), 256, 0, stream>>>(v_c, kp_s, kptv_s, Tt, Ee, Mm);
    kptv_wmma_kernel<<<dim3(Mm / 128, Ee / 128, Bb), 256, 0, stream>>>(v_s, kp_c, kptv_c, Tt, Ee, Mm);

    // 7) D = qp . ksum per row
    rowdot_kernel<<<(unsigned)(Rr / 8), 256, 0, stream>>>(qp_s, ksum_s, Dsv, Mm, Tt);
    rowdot_kernel<<<(unsigned)(Rr / 8), 256, 0, stream>>>(qp_c, ksum_c, Dcv, Mm, Tt);

    // 8) y = (qp @ kptv^T) / (D + eps)   (per-batch weight)
    gemm(2, qp_s, kptv_s, ynum_s, nullptr, Dsv, nullptr, Tt, Ee, Mm, Bb, (long)Ee * Mm, 0.f);
    gemm(2, qp_c, kptv_c, ynum_c, nullptr, Dcv, nullptr, Tt, Ee, Mm, Bb, (long)Ee * Mm, 0.f);

    // 9) s_pre = input + proj(y) + bias   (f32)
    gemm(3, ynum_s, h_ps, s_pre, (const float*)d_in[15], nullptr, saliency, (int)Rr, Ee, Ee, 1, 0, 0.f);
    gemm(3, ynum_c, h_pc, c_pre, (const float*)d_in[17], nullptr, contour, (int)Rr, Ee, Ee, 1, 0, 0.f);

    // 10) LayerNorm 2 -> f16
    layernorm_f16_kernel<<<(unsigned)Rr, 256, 0, stream>>>(
        s_pre, (const float*)d_in[22], (const float*)d_in[23], ln2s, Ee);
    layernorm_f16_kernel<<<(unsigned)Rr, 256, 0, stream>>>(
        c_pre, (const float*)d_in[24], (const float*)d_in[25], ln2c, Ee);

    // 11) MLP first layer with exact GELU
    gemm(4, ln2s, h_m1s, gel_s, (const float*)d_in[27], nullptr, nullptr, (int)Rr, Ee, Ee, 1, 0, 0.f);
    gemm(4, ln2c, h_m1c, gel_c, (const float*)d_in[31], nullptr, nullptr, (int)Rr, Ee, Ee, 1, 0, 0.f);

    // 12) MLP second layer + residual -> d_out (f32)
    float* outS = (float*)d_out;
    float* outC = outS + Rr * Ee;
    gemm(3, gel_s, h_m2s, outS, (const float*)d_in[29], nullptr, s_pre, (int)Rr, Ee, Ee, 1, 0, 0.f);
    gemm(3, gel_c, h_m2c, outC, (const float*)d_in[33], nullptr, c_pre, (int)Rr, Ee, Ee, 1, 0, 0.f);
}
